// LightNetAttention_82437602279424
// MI455X (gfx1250) — compile-verified, matrix operand order checked by reading, not run on hardware
//
#include <hip/hip_runtime.h>

// ---------------------------------------------------------------------------
// LightNet attention for MI455X (gfx1250, wave32, WMMA)
// GEMMs: v_wmma_f32_16x16x32_bf16, LDS double-buffered tiles fed by the
// CDNA5 async global->LDS engine (ASYNCcnt) when available.
// ---------------------------------------------------------------------------

#define SEQ_N   4096
#define HDIM    1024
#define NHEADS  16
#define HEAD_D  64
#define NBATCH  4
#define TOKENS  (NBATCH * SEQ_N)   // 16384

#define BLK_M   128
#define BLK_N   128
#define BLK_K   32
#define LDS_STRIDE 40              // 32 + 8 pad (80B rows, 16B aligned)

typedef __bf16 bf16_t;
typedef bf16_t v16bf __attribute__((ext_vector_type(16)));
typedef bf16_t v8bf  __attribute__((ext_vector_type(8)));
typedef float  v8f   __attribute__((ext_vector_type(8)));
typedef int    v4i   __attribute__((ext_vector_type(4)));

#if defined(__AMDGCN__) && __has_builtin(__builtin_amdgcn_global_load_async_to_lds_b128)
#define ASYNC_LDS 1
typedef __attribute__((address_space(1))) v4i* gas_v4i;
typedef __attribute__((address_space(3))) v4i* las_v4i;
#else
#define ASYNC_LDS 0
#endif

__device__ __forceinline__ void wait_async0() {
#if defined(__AMDGCN__)
#if __has_builtin(__builtin_amdgcn_s_wait_asynccnt)
  __builtin_amdgcn_s_wait_asynccnt(0);
#else
  asm volatile("s_wait_asynccnt 0x0" ::: "memory");
#endif
#endif
}

__device__ __forceinline__ bf16_t f2bf(float f) {
  union { float f; unsigned u; } a; a.f = f;
  unsigned u = a.u;
  unsigned r = u + 0x7fffu + ((u >> 16) & 1u);   // round-to-nearest-even
  unsigned short h = (unsigned short)(r >> 16);
  bf16_t o;
  __builtin_memcpy(&o, &h, sizeof(o));
  return o;
}

// Load one 16x32 bf16 A/B fragment from a row-major [rows, LDS_STRIDE] LDS
// tile. Per ISA layout: lane holds row (lane%16); element e covers
// K = (e/8)*16 + (lane/16)*8 + (e%8)  -> two contiguous 16-byte ds loads.
__device__ __forceinline__ v16bf load_frag_lds(const bf16_t* base, int row, int half) {
  const bf16_t* p = base + row * LDS_STRIDE + half * 8;
  v8bf lo = *reinterpret_cast<const v8bf*>(p);
  v8bf hi = *reinterpret_cast<const v8bf*>(p + 16);
  v16bf r;
#pragma unroll
  for (int e = 0; e < 8; ++e) { r[e] = lo[e]; r[e + 8] = hi[e]; }
  return r;
}

// Stage a BLK x 32 bf16 tile (global, row-major, leading dim ld) into LDS.
// 512 chunks of 16B; 256 threads x 2 chunks. Rows clamped for edge tiles.
__device__ __forceinline__ void load_tile(const bf16_t* gbase, int rows, int ld,
                                          int row0, int k0, bf16_t* lbase, int tid) {
#pragma unroll
  for (int t = 0; t < 2; ++t) {
    const int c  = tid + t * 256;        // chunk id 0..511
    const int r  = c >> 2;               // tile row 0..127
    const int cc = (c & 3) * 8;          // k chunk 0/8/16/24
    int gr = row0 + r;
    if (gr > rows - 1) gr = rows - 1;
    const bf16_t* gp = gbase + (long long)gr * ld + k0 + cc;
    bf16_t* lp = lbase + r * LDS_STRIDE + cc;
#if ASYNC_LDS
    __builtin_amdgcn_global_load_async_to_lds_b128((gas_v4i)gp, (las_v4i)lp, 0, 0);
#else
    *reinterpret_cast<v8bf*>(lp) = *reinterpret_cast<const v8bf*>(gp);
#endif
  }
}

// ---------------------------------------------------------------------------
// Generic batched WMMA GEMM:  C[m,n] = sum_k A[m,k] * W[n,k]  (+ bias[n])
// 256 threads = 8 waves as 4(M) x 2(N); block tile 128x128; wave tile 32x64.
// K loop in steps of 32, double-buffered LDS tiles via async-to-LDS.
// Epilogue modes:
//   0: store f32 to Cf
//   2: store bf16 to Cbf
//   3: store bf16( sigmoid(c) * aux[same index] ) to Cbf      (gate fusion)
//   4: qkv scatter: silu(q)->qbf[b,h,n,d], k->kT_f32[b,h,d,n], v->vT_bf[b,h,d,n]
// ---------------------------------------------------------------------------
struct GemmP {
  const bf16_t* A; const bf16_t* W; const float* bias;
  float* Cf; bf16_t* Cbf; const float* aux;
  float* kT; bf16_t* vT; bf16_t* qbf;
  int M, Nn, K, lda, ldw, ldc;
  int heads;
  long long sAb, sAh, sWb, sWh, sCb, sCh;
  int mode;
};

__global__ __launch_bounds__(256) void gemm_bf16_wmma(GemmP p) {
  __shared__ bf16_t sA[2][BLK_M * LDS_STRIDE];
  __shared__ bf16_t sB[2][BLK_N * LDS_STRIDE];

  const int tid  = threadIdx.x;
  const int lane = tid & 31;
  const int wave = tid >> 5;
  const int wm = wave >> 1;          // 0..3 -> 32-row slice
  const int wn = wave & 1;           // 0..1 -> 64-col slice
  const int half = lane >> 4;        // 0..1
  const int lr = lane & 15;          // 0..15

  const int z = blockIdx.z;
  const int b = z / p.heads;
  const int h = z - b * p.heads;
  const bf16_t* A = p.A + b * p.sAb + h * p.sAh;
  const bf16_t* W = p.W + b * p.sWb + h * p.sWh;
  const long long coff = b * p.sCb + h * p.sCh;

  const int mblk = blockIdx.y * BLK_M;
  const int nblk = blockIdx.x * BLK_N;

  v8f acc[2][4];
#pragma unroll
  for (int i = 0; i < 2; ++i)
#pragma unroll
    for (int j = 0; j < 4; ++j) acc[i][j] = {};

  // Prologue: stage first K tile.
  load_tile(A, p.M, p.lda, mblk, 0, sA[0], tid);
  load_tile(W, p.Nn, p.ldw, nblk, 0, sB[0], tid);
  wait_async0();
  __syncthreads();

  for (int k0 = 0; k0 < p.K; k0 += BLK_K) {
    const int cur = (k0 >> 5) & 1;
    if (k0 + BLK_K < p.K) {
      load_tile(A, p.M, p.lda, mblk, k0 + BLK_K, sA[cur ^ 1], tid);
      load_tile(W, p.Nn, p.ldw, nblk, k0 + BLK_K, sB[cur ^ 1], tid);
    }

    v16bf af[2];
#pragma unroll
    for (int i = 0; i < 2; ++i)
      af[i] = load_frag_lds(sA[cur], wm * 32 + i * 16 + lr, half);
    v16bf wf[4];
#pragma unroll
    for (int j = 0; j < 4; ++j)
      wf[j] = load_frag_lds(sB[cur], wn * 64 + j * 16 + lr, half);

#pragma unroll
    for (int i = 0; i < 2; ++i)
#pragma unroll
      for (int j = 0; j < 4; ++j)
        acc[i][j] = __builtin_amdgcn_wmma_f32_16x16x32_bf16(
            false, af[i], false, wf[j], (short)0, acc[i][j], false, false);

    wait_async0();       // prefetch landed
    __syncthreads();     // everyone done reading 'cur'; next tile visible
  }

  // Epilogue: C/D layout -> element r of lane = (row +half*8+r, col lr)
#pragma unroll
  for (int i = 0; i < 2; ++i) {
#pragma unroll
    for (int j = 0; j < 4; ++j) {
#pragma unroll
      for (int r = 0; r < 8; ++r) {
        const int m = mblk + wm * 32 + i * 16 + half * 8 + r;
        const int n = nblk + wn * 64 + j * 16 + lr;
        if (m >= p.M || n >= p.Nn) continue;
        float c = acc[i][j][r];
        if (p.bias) c += p.bias[n];
        if (p.mode == 0) {
          p.Cf[coff + (long long)m * p.ldc + n] = c;
        } else if (p.mode == 2) {
          p.Cbf[coff + (long long)m * p.ldc + n] = f2bf(c);
        } else if (p.mode == 3) {
          const float g = 1.0f / (1.0f + __expf(-c));
          const float v = p.aux[coff + (long long)m * p.ldc + n];
          p.Cbf[coff + (long long)m * p.ldc + n] = f2bf(g * v);
        } else {  // mode 4: qkv split/scatter
          const int sect = n / HDIM;          // 0=q 1=k 2=v
          const int cc = n - sect * HDIM;
          const int head = cc / HEAD_D;
          const int d = cc - head * HEAD_D;
          const int bb = m / SEQ_N;
          const int nn = m - bb * SEQ_N;
          const long long bh = (long long)(bb * NHEADS + head);
          if (sect == 0) {
            const float s = c / (1.0f + __expf(-c));   // silu
            p.qbf[(bh * SEQ_N + nn) * HEAD_D + d] = f2bf(s);
          } else if (sect == 1) {
            p.kT[(bh * HEAD_D + d) * SEQ_N + nn] = c;
          } else {
            p.vT[(bh * HEAD_D + d) * SEQ_N + nn] = f2bf(c);
          }
        }
      }
    }
  }
}

// ---------------------------------------------------------------------------
// f32 -> bf16 elementwise convert
// ---------------------------------------------------------------------------
__global__ __launch_bounds__(256) void cvt_bf16(const float* __restrict__ in,
                                                bf16_t* __restrict__ out,
                                                long long n) {
  long long i = (long long)blockIdx.x * blockDim.x + threadIdx.x;
  const long long stride = (long long)gridDim.x * blockDim.x;
  for (; i < n; i += stride) out[i] = f2bf(in[i]);
}

// ---------------------------------------------------------------------------
// Softmax over sequence dim: kT_f32[row, 0..SEQ_N) -> kT_bf (rows = B*H*64)
// ---------------------------------------------------------------------------
__global__ __launch_bounds__(256) void softmax_rows(const float* __restrict__ kT,
                                                    bf16_t* __restrict__ kTbf) {
  __shared__ float red[256];
  const long long row = blockIdx.x;
  const float* x = kT + row * (long long)SEQ_N;
  const int t = threadIdx.x;

  float mx = -1e30f;
  for (int i = t; i < SEQ_N; i += 256) mx = fmaxf(mx, x[i]);
  red[t] = mx; __syncthreads();
  for (int s = 128; s > 0; s >>= 1) {
    if (t < s) red[t] = fmaxf(red[t], red[t + s]);
    __syncthreads();
  }
  mx = red[0]; __syncthreads();

  float sum = 0.0f;
  for (int i = t; i < SEQ_N; i += 256) sum += __expf(x[i] - mx);
  red[t] = sum; __syncthreads();
  for (int s = 128; s > 0; s >>= 1) {
    if (t < s) red[t] += red[t + s];
    __syncthreads();
  }
  const float inv = 1.0f / red[0];

  bf16_t* o = kTbf + row * (long long)SEQ_N;
  for (int i = t; i < SEQ_N; i += 256) o[i] = f2bf(__expf(x[i] - mx) * inv);
}

// ---------------------------------------------------------------------------
// LayerNorm in place over last dim (rows = TOKENS, width = HDIM)
// ---------------------------------------------------------------------------
__global__ __launch_bounds__(256) void layernorm_rows(float* __restrict__ xio,
                                                      const float* __restrict__ g,
                                                      const float* __restrict__ bb) {
  __shared__ float red[256];
  const long long row = blockIdx.x;
  float* x = xio + row * (long long)HDIM;
  const int t = threadIdx.x;

  float s = 0.0f;
  for (int i = t; i < HDIM; i += 256) s += x[i];
  red[t] = s; __syncthreads();
  for (int k = 128; k > 0; k >>= 1) { if (t < k) red[t] += red[t + k]; __syncthreads(); }
  const float mean = red[0] * (1.0f / HDIM); __syncthreads();

  float v = 0.0f;
  for (int i = t; i < HDIM; i += 256) { const float d = x[i] - mean; v += d * d; }
  red[t] = v; __syncthreads();
  for (int k = 128; k > 0; k >>= 1) { if (t < k) red[t] += red[t + k]; __syncthreads(); }
  const float inv = rsqrtf(red[0] * (1.0f / HDIM) + 1e-5f);

  for (int i = t; i < HDIM; i += 256)
    x[i] = (x[i] - mean) * inv * g[i] + bb[i];
}

// ---------------------------------------------------------------------------
// Host-side orchestration
// ---------------------------------------------------------------------------
static inline void launch_gemm(const GemmP& p, int zdim, hipStream_t stream) {
  dim3 grid((p.Nn + BLK_N - 1) / BLK_N, (p.M + BLK_M - 1) / BLK_M, zdim);
  gemm_bf16_wmma<<<grid, dim3(256), 0, stream>>>(p);
}

extern "C" void kernel_launch(void* const* d_in, const int* in_sizes, int n_in,
                              void* d_out, int out_size, void* d_ws, size_t ws_size,
                              hipStream_t stream) {
  const float* x_f   = (const float*)d_in[0];
  const float* w_qkv = (const float*)d_in[1];
  const float* b_qkv = (const float*)d_in[2];
  const float* w_g1  = (const float*)d_in[3];
  const float* b_g1  = (const float*)d_in[4];
  const float* w_g2  = (const float*)d_in[5];
  const float* b_g2  = (const float*)d_in[6];
  const float* w_out = (const float*)d_in[7];
  const float* b_out = (const float*)d_in[8];
  const float* ln_g  = (const float*)d_in[9];
  const float* ln_b  = (const float*)d_in[10];

  char* ws = (char*)d_ws;
  size_t off = 0;
  auto alloc = [&](size_t bytes) -> void* {
    void* p = ws + off;
    off = (off + bytes + 255) & ~(size_t)255;
    return p;
  };

  bf16_t* x_bf     = (bf16_t*)alloc((size_t)TOKENS * HDIM * 2);
  bf16_t* w_qkv_bf = (bf16_t*)alloc((size_t)3 * HDIM * HDIM * 2);
  bf16_t* w_g1_bf  = (bf16_t*)alloc((size_t)HEAD_D * HDIM * 2);
  bf16_t* w_g2_bf  = (bf16_t*)alloc((size_t)HDIM * HEAD_D * 2);
  bf16_t* w_out_bf = (bf16_t*)alloc((size_t)HDIM * HDIM * 2);
  bf16_t* q_bf     = (bf16_t*)alloc((size_t)NBATCH * NHEADS * SEQ_N * HEAD_D * 2);
  float*  kT_f32   = (float*) alloc((size_t)NBATCH * NHEADS * HEAD_D * SEQ_N * 4);
  bf16_t* kT_bf    = (bf16_t*)alloc((size_t)NBATCH * NHEADS * HEAD_D * SEQ_N * 2);
  bf16_t* vT_bf    = (bf16_t*)alloc((size_t)NBATCH * NHEADS * HEAD_D * SEQ_N * 2);
  bf16_t* kvT_bf   = (bf16_t*)alloc((size_t)NBATCH * NHEADS * HEAD_D * HEAD_D * 2);
  float*  attn_f32 = (float*) alloc((size_t)TOKENS * HDIM * 4);
  bf16_t* gh_bf    = (bf16_t*)alloc((size_t)TOKENS * HEAD_D * 2);
  bf16_t* gated_bf = (bf16_t*)alloc((size_t)TOKENS * HDIM * 2);

  const dim3 blk(256);

  // ---- f32 -> bf16 conversions -------------------------------------------
  cvt_bf16<<<2048, blk, 0, stream>>>(x_f,   x_bf,     (long long)TOKENS * HDIM);
  cvt_bf16<<<1024, blk, 0, stream>>>(w_qkv, w_qkv_bf, (long long)3 * HDIM * HDIM);
  cvt_bf16<<<64,   blk, 0, stream>>>(w_g1,  w_g1_bf,  (long long)HEAD_D * HDIM);
  cvt_bf16<<<64,   blk, 0, stream>>>(w_g2,  w_g2_bf,  (long long)HDIM * HEAD_D);
  cvt_bf16<<<512,  blk, 0, stream>>>(w_out, w_out_bf, (long long)HDIM * HDIM);

  // ---- qkv projection + head split / silu / transpose scatter ------------
  {
    GemmP p{};
    p.A = x_bf; p.W = w_qkv_bf; p.bias = b_qkv;
    p.kT = kT_f32; p.vT = vT_bf; p.qbf = q_bf;
    p.M = TOKENS; p.Nn = 3 * HDIM; p.K = HDIM;
    p.lda = HDIM; p.ldw = HDIM; p.ldc = 0;
    p.heads = 1; p.mode = 4;
    launch_gemm(p, 1, stream);
  }

  // ---- softmax over sequence dim (rows = B*H*64, each SEQ_N long) ---------
  softmax_rows<<<NBATCH * NHEADS * HEAD_D, blk, 0, stream>>>(kT_f32, kT_bf);

  // ---- kv^T[e,d] = sum_n vT[e,n] * kT[d,n]   (per b,h) --------------------
  {
    GemmP p{};
    p.A = vT_bf; p.W = kT_bf; p.bias = nullptr;
    p.Cbf = kvT_bf;
    p.M = HEAD_D; p.Nn = HEAD_D; p.K = SEQ_N;
    p.lda = SEQ_N; p.ldw = SEQ_N; p.ldc = HEAD_D;
    p.heads = NHEADS;
    p.sAb = (long long)NHEADS * HEAD_D * SEQ_N; p.sAh = (long long)HEAD_D * SEQ_N;
    p.sWb = p.sAb;                              p.sWh = p.sAh;
    p.sCb = (long long)NHEADS * HEAD_D * HEAD_D; p.sCh = (long long)HEAD_D * HEAD_D;
    p.mode = 2;
    launch_gemm(p, NBATCH * NHEADS, stream);
  }

  // ---- attn[b,n,h*64+e] = sum_d q[b,h,n,d] * kvT[e,d] ---------------------
  {
    GemmP p{};
    p.A = q_bf; p.W = kvT_bf; p.bias = nullptr;
    p.Cf = attn_f32;
    p.M = SEQ_N; p.Nn = HEAD_D; p.K = HEAD_D;
    p.lda = HEAD_D; p.ldw = HEAD_D; p.ldc = HDIM;
    p.heads = NHEADS;
    p.sAb = (long long)NHEADS * SEQ_N * HEAD_D; p.sAh = (long long)SEQ_N * HEAD_D;
    p.sWb = (long long)NHEADS * HEAD_D * HEAD_D; p.sWh = (long long)HEAD_D * HEAD_D;
    p.sCb = (long long)SEQ_N * HDIM;            p.sCh = HEAD_D;
    p.mode = 0;
    launch_gemm(p, NBATCH * NHEADS, stream);
  }

  // ---- layernorm (in place on attn_f32) -----------------------------------
  layernorm_rows<<<TOKENS, blk, 0, stream>>>(attn_f32, ln_g, ln_b);

  // ---- gate hidden: x @ w_g1^T + b_g1  -> bf16 [TOKENS, 64] ---------------
  {
    GemmP p{};
    p.A = x_bf; p.W = w_g1_bf; p.bias = b_g1;
    p.Cbf = gh_bf;
    p.M = TOKENS; p.Nn = HEAD_D; p.K = HDIM;
    p.lda = HDIM; p.ldw = HDIM; p.ldc = HEAD_D;
    p.heads = 1; p.mode = 2;
    launch_gemm(p, 1, stream);
  }

  // ---- gate: sigmoid(gh @ w_g2^T + b_g2) * ln  -> bf16 gated --------------
  {
    GemmP p{};
    p.A = gh_bf; p.W = w_g2_bf; p.bias = b_g2;
    p.Cbf = gated_bf; p.aux = attn_f32;
    p.M = TOKENS; p.Nn = HDIM; p.K = HEAD_D;
    p.lda = HEAD_D; p.ldw = HEAD_D; p.ldc = HDIM;
    p.heads = 1; p.mode = 3;
    launch_gemm(p, 1, stream);
  }

  // ---- output projection: gated @ w_out^T + b_out -> d_out (f32) ----------
  {
    GemmP p{};
    p.A = gated_bf; p.W = w_out_bf; p.bias = b_out;
    p.Cf = (float*)d_out;
    p.M = TOKENS; p.Nn = HDIM; p.K = HDIM;
    p.lda = HDIM; p.ldw = HDIM; p.ldc = HDIM;
    p.heads = 1; p.mode = 0;
    launch_gemm(p, 1, stream);
  }

  (void)in_sizes; (void)n_in; (void)out_size; (void)ws_size;
}
